// MultiHeadAttention_59219009077555
// MI455X (gfx1250) — matrix operand order repaired and checked by reading, not compiled, not run on hardware
//
#include <hip/hip_runtime.h>
#include <hip/hip_bf16.h>

#define B_ 4
#define T_ 2048
#define D_ 1024
#define H_ 16
#define HD_ 64
#define M_ (B_*T_)   // 8192

typedef __attribute__((ext_vector_type(16))) _Float16 v16h;
typedef __attribute__((ext_vector_type(8)))  _Float16 v8h;
typedef __attribute__((ext_vector_type(4)))  _Float16 v4h;
typedef __attribute__((ext_vector_type(8)))  float    v8f;

union V16 { v16h v; v8h h[2]; };

__device__ __forceinline__ v16h ldv16(const _Float16* p0, const _Float16* p1) {
  V16 u; u.h[0] = *(const v8h*)p0; u.h[1] = *(const v8h*)p1; return u.v;
}

#define WMMA_F16(a,b,c) \
  __builtin_amdgcn_wmma_f32_16x16x32_f16(false,(a),false,(b),(short)0,(c),false,false)

// ---- CDNA5 async global->LDS copy (ASYNCcnt-tracked), with fallback ----
#if defined(__gfx1250__) && __has_builtin(__builtin_amdgcn_global_load_async_to_lds_b128)
#define HAS_ASYNC_LDS 1
#else
#define HAS_ASYNC_LDS 0
#endif

// builtin parameter type: pointer to gcc-style int4 vector, AS1 (global) src
// and AS3 (LDS) dst, plus immediate offset + cpol.
typedef int gv4i __attribute__((__vector_size__(16)));
typedef __attribute__((address_space(1))) gv4i* as1_v4i;
typedef __attribute__((address_space(3))) gv4i* as3_v4i;

__device__ __forceinline__ void cp16_g2l(const _Float16* g, _Float16* l) {
#if HAS_ASYNC_LDS
  __builtin_amdgcn_global_load_async_to_lds_b128(
      (as1_v4i)(_Float16*)g, (as3_v4i)l, 0, 0);
#else
  *(v8h*)l = *(const v8h*)g;
#endif
}

// Wait until at most N async copies remain in flight (in-order completion).
template <int N>
__device__ __forceinline__ void wait_async() {
#if HAS_ASYNC_LDS
#if __has_builtin(__builtin_amdgcn_s_wait_asynccnt)
  __builtin_amdgcn_s_wait_asynccnt(N);
#else
  asm volatile("s_wait_asynccnt %0" :: "i"(N) : "memory");
#endif
#endif
}

// ---------------- Phase 0: f32 -> f16 conversion ----------------
__global__ void cvt_f32_to_f16(const float* __restrict__ src,
                               _Float16* __restrict__ dst, int n4) {
  int i = blockIdx.x * blockDim.x + threadIdx.x;
  if (i >= n4) return;
  float4 f = ((const float4*)src)[i];
  v4h h;
  h[0] = (_Float16)f.x; h[1] = (_Float16)f.y;
  h[2] = (_Float16)f.z; h[3] = (_Float16)f.w;
  ((v4h*)dst)[i] = h;
}

// ---------------- Phase 1/3: C = A[M,K] * W[N,K]^T + bias --------
// mode 0: f32 row-major out; mode 1: f16 row-major out;
// mode 2: f16 out scattered as Vt[B,H,HD,T] (value projection).
__global__ void __launch_bounds__(256) gemm_bt_f16(
    const _Float16* __restrict__ A, const _Float16* __restrict__ W,
    const float* __restrict__ bias, float* __restrict__ outF,
    _Float16* __restrict__ outH, int M, int N, int K, int mode)
{
  const int lane  = threadIdx.x & 31;
  const int wid   = blockIdx.x * 8 + (threadIdx.x >> 5);
  const int lh    = lane & 15;
  const int sel8  = (lane & 16) ? 8  : 0;   // A-operand K shift / C row shift
  const int sel16 = (lane & 16) ? 16 : 0;   // unused (B-operand K shift is sel8, line 95)
  const int NT = N >> 6;
  const int m0 = (wid / NT) * 32;
  const int n0 = (wid % NT) * 64;

  v8f acc[2][4] = {};

  const _Float16* a0p = A + (size_t)(m0 +      lh) * K;
  const _Float16* a1p = A + (size_t)(m0 + 16 + lh) * K;
  const _Float16* wp[4];
  #pragma unroll
  for (int g = 0; g < 4; ++g)
    wp[g] = W + (size_t)(n0 + g*16 + lh) * K + sel8;

  #pragma unroll 4
  for (int k0 = 0; k0 < K; k0 += 32) {
    v16h a0 = ldv16(a0p + k0 + sel8, a0p + k0 + 16 + sel8);
    v16h a1 = ldv16(a1p + k0 + sel8, a1p + k0 + 16 + sel8);
    #pragma unroll
    for (int g = 0; g < 4; ++g) {
      v16h w = ldv16(wp[g] + k0, wp[g] + k0 + 16);
      acc[0][g] = WMMA_F16(a0, w, acc[0][g]);
      acc[1][g] = WMMA_F16(a1, w, acc[1][g]);
    }
  }

  #pragma unroll
  for (int g = 0; g < 4; ++g) {
    const int col = n0 + g*16 + lh;
    const float bn = bias[col];
    #pragma unroll
    for (int i = 0; i < 2; ++i) {
      #pragma unroll
      for (int r = 0; r < 8; ++r) {
        const int row = m0 + i*16 + r + sel8;
        const float val = acc[i][g][r] + bn;
        if (mode == 0) {
          outF[(size_t)row * N + col] = val;
        } else if (mode == 1) {
          outH[(size_t)row * N + col] = (_Float16)val;
        } else {
          const int h = col >> 6, hd = col & 63;
          const int b = row >> 11, t = row & 2047;
          outH[(size_t)((b*H_ + h)*HD_ + hd) * T_ + t] = (_Float16)val;
        }
      }
    }
  }
}

// ---------------- Phase 2: block flash attention -----------------
// Q,K: f16 [M,D] (head h at col h*64). Vt: f16 [B,H,HD,T]. O: f16 [M,D].
// Block = 4 waves = 64 q rows. K/V 64x64 tiles double-buffered in LDS via
// async global->LDS copies: next tile prefetches while current computes;
// s_wait_asynccnt 8 retires only the previous tile's 8 copies (in-order).
#define LSTRIDE 72   // padded row stride in halves (bank-conflict avoidance)

__global__ void __launch_bounds__(128) flash_attn_f16(
    const _Float16* __restrict__ Q, const _Float16* __restrict__ Kh,
    const _Float16* __restrict__ Vt, const unsigned char* __restrict__ mask,
    _Float16* __restrict__ O)
{
  __shared__ _Float16 ktile[2][64*LSTRIDE];  // 64 keys x 64 d, double-buffered
  __shared__ _Float16 vtile[2][64*LSTRIDE];  // 64 hd x 64 t, double-buffered
  __shared__ _Float16 plds[4][16*LSTRIDE];   // per-wave P transpose tile

  const int wave  = threadIdx.x >> 5;
  const int lane  = threadIdx.x & 31;
  const int lh    = lane & 15;
  const int sel8  = (lane & 16) ? 8  : 0;
  const int sel16 = (lane & 16) ? 16 : 0;
  const int q0 = blockIdx.x * 64 + wave * 16;
  const int bh = blockIdx.y;
  const int b = bh >> 4, h = bh & 15;

  // Q A-operands (16 rows x 64 K), pre-scaled by 1/sqrt(HD)=0.125
  const _Float16* qrow = Q + (size_t)(b*T_ + q0 + lh) * D_ + h*HD_;
  v16h aq[2];
  #pragma unroll
  for (int c = 0; c < 2; ++c) {
    aq[c] = ldv16(qrow + c*32 + sel8, qrow + c*32 + 16 + sel8);
    #pragma unroll
    for (int i = 0; i < 16; ++i) aq[c][i] = aq[c][i] * (_Float16)0.125f;
  }

  v8f oacc[4] = {};
  float mrow[8], lrow[8];
  #pragma unroll
  for (int r = 0; r < 8; ++r) { mrow[r] = -1.0e30f; lrow[r] = 0.0f; }

  _Float16* pl = &plds[wave][0];
  const unsigned char* mbase = mask + (size_t)(b*T_ + q0) * T_;
  const _Float16* kbase = Kh + (size_t)(b*T_) * D_ + h*HD_;
  const _Float16* vbase = Vt + (size_t)(bh*HD_) * T_;

  // cooperative-copy coordinates: each wave stages rows [wave*16, wave*16+16)
  const int crow = wave*16 + (lane >> 3);   // base row; +4 per issue
  const int cch  = (lane & 7) * 8;          // 8-half (16B) chunk within row

  // stage one 64-key K/V tile pair into buffer p (8 async b128 per wave)
  auto stage = [&](int kt, int p) {
    #pragma unroll
    for (int i = 0; i < 4; ++i) {
      const int r = crow + i*4;
      cp16_g2l(kbase + (size_t)(kt + r)*D_ + cch, &ktile[p][r*LSTRIDE + cch]);
      cp16_g2l(vbase + (size_t)r*T_ + kt + cch,   &vtile[p][r*LSTRIDE + cch]);
    }
  };

  stage(0, 0);
  int p = 0;
  for (int kt = 0; kt < T_; kt += 64) {
    // prefetch next tile into the other buffer while we compute this one
    const bool pre = (kt + 64 < T_);
    if (pre) {
      stage(kt + 64, p ^ 1);
      wait_async<8>();   // retire previous 8 copies; keep prefetch in flight
    } else {
      wait_async<0>();
    }
    __syncthreads();

    // ---- S = Q * K^T, 4 groups of 16 keys (8 WMMAs), B from LDS ----
    v8f s[4];
    #pragma unroll
    for (int g = 0; g < 4; ++g) {
      const _Float16* kr = &ktile[p][(g*16 + lh) * LSTRIDE];
      v16h k0 = ldv16(kr + sel8,      kr + 16 + sel8);
      v16h k1 = ldv16(kr + 32 + sel8, kr + 48 + sel8);
      v8f z = {};
      z    = WMMA_F16(aq[0], k0, z);
      s[g] = WMMA_F16(aq[1], k1, z);
    }
    // ---- online softmax per row (rows split by lane half) ----
    #pragma unroll
    for (int r = 0; r < 8; ++r) {
      const int qm = r + sel8;
      float x[4];
      #pragma unroll
      for (int g = 0; g < 4; ++g) {
        const unsigned char mv = mbase[(size_t)qm * T_ + kt + g*16 + lh];
        x[g] = mv ? -3.0e38f : s[g][r];
      }
      float mx = fmaxf(fmaxf(x[0], x[1]), fmaxf(x[2], x[3]));
      mx = fmaxf(mx, __shfl_xor(mx, 1, 32));
      mx = fmaxf(mx, __shfl_xor(mx, 2, 32));
      mx = fmaxf(mx, __shfl_xor(mx, 4, 32));
      mx = fmaxf(mx, __shfl_xor(mx, 8, 32));
      const float mnew  = fmaxf(mrow[r], mx);
      const float alpha = exp2f((mrow[r] - mnew) * 1.44269504f);
      mrow[r] = mnew;
      float ps = 0.0f;
      #pragma unroll
      for (int g = 0; g < 4; ++g) {
        const float pv = exp2f((x[g] - mnew) * 1.44269504f);
        ps += pv;
        pl[qm*LSTRIDE + g*16 + lh] = (_Float16)pv;   // stash P for transpose
      }
      ps += __shfl_xor(ps, 1, 32);
      ps += __shfl_xor(ps, 2, 32);
      ps += __shfl_xor(ps, 4, 32);
      ps += __shfl_xor(ps, 8, 32);
      lrow[r] = lrow[r] * alpha + ps;
      #pragma unroll
      for (int od = 0; od < 4; ++od) oacc[od][r] *= alpha;
    }
    // ---- reload P in A-operand layout from wave-private LDS ----
    v16h ap0 = ldv16(pl + lh*LSTRIDE + sel8,      pl + lh*LSTRIDE + 16 + sel8);
    v16h ap1 = ldv16(pl + lh*LSTRIDE + 32 + sel8, pl + lh*LSTRIDE + 48 + sel8);
    // ---- O += P * V, B-operand from LDS (V pre-transposed) ----
    #pragma unroll
    for (int od = 0; od < 4; ++od) {
      const _Float16* vr = &vtile[p][(od*16 + lh) * LSTRIDE];
      v16h v0 = ldv16(vr + sel8,      vr + 16 + sel8);
      v16h v1 = ldv16(vr + 32 + sel8, vr + 48 + sel8);
      oacc[od] = WMMA_F16(ap0, v0, oacc[od]);
      oacc[od] = WMMA_F16(ap1, v1, oacc[od]);
    }
    __syncthreads();   // buffer p fully consumed before it is restaged
    p ^= 1;
  }

  #pragma unroll
  for (int r = 0; r < 8; ++r) {
    const float rinv = (lrow[r] > 0.0f) ? (1.0f / lrow[r]) : 0.0f;
    #pragma unroll
    for (int od = 0; od < 4; ++od) {
      O[(size_t)(b*T_ + q0 + r + sel8) * D_ + h*HD_ + od*16 + lh] =
          (_Float16)(oacc[od][r] * rinv);
    }
  }
}

// ---------------- Launch ----------------
extern "C" void kernel_launch(void* const* d_in, const int* in_sizes, int n_in,
                              void* d_out, int out_size, void* d_ws, size_t ws_size,
                              hipStream_t stream) {
  (void)in_sizes; (void)n_in; (void)out_size; (void)ws_size;
  const float* q  = (const float*)d_in[0];
  const float* k  = (const float*)d_in[1];
  const float* v  = (const float*)d_in[2];
  const unsigned char* mask = (const unsigned char*)d_in[3];
  const float* Wq = (const float*)d_in[4];
  const float* bq = (const float*)d_in[5];
  const float* Wk = (const float*)d_in[6];
  const float* bk = (const float*)d_in[7];
  const float* Wv = (const float*)d_in[8];
  const float* bv = (const float*)d_in[9];
  const float* Wo = (const float*)d_in[10];
  const float* bo = (const float*)d_in[11];
  float* out = (float*)d_out;

  // Workspace layout (f16 elements). Total = 6*MD + 4*D*D halves ~= 104 MiB.
  _Float16* ws = (_Float16*)d_ws;
  const size_t MD  = (size_t)M_ * D_;
  const size_t DSQ = (size_t)D_ * D_;
  _Float16* Xq  = ws;
  _Float16* Xk  = Xq  + MD;
  _Float16* Xv  = Xk  + MD;
  _Float16* Qh  = Xv  + MD;
  _Float16* Khb = Qh  + MD;
  _Float16* Vtb = Khb + MD;
  _Float16* Wqh = Vtb + MD;
  _Float16* Wkh = Wqh + DSQ;
  _Float16* Wvh = Wkh + DSQ;
  _Float16* Woh = Wvh + DSQ;
  _Float16* Oh  = Xq;   // Xq dead after Q projection -> reuse for attention out

  const int cb = 256;
  cvt_f32_to_f16<<<(int)(MD / 4 / cb), cb, 0, stream>>>(q,  Xq,  (int)(MD / 4));
  cvt_f32_to_f16<<<(int)(MD / 4 / cb), cb, 0, stream>>>(k,  Xk,  (int)(MD / 4));
  cvt_f32_to_f16<<<(int)(MD / 4 / cb), cb, 0, stream>>>(v,  Xv,  (int)(MD / 4));
  cvt_f32_to_f16<<<(int)(DSQ / 4 / cb), cb, 0, stream>>>(Wq, Wqh, (int)(DSQ / 4));
  cvt_f32_to_f16<<<(int)(DSQ / 4 / cb), cb, 0, stream>>>(Wk, Wkh, (int)(DSQ / 4));
  cvt_f32_to_f16<<<(int)(DSQ / 4 / cb), cb, 0, stream>>>(Wv, Wvh, (int)(DSQ / 4));
  cvt_f32_to_f16<<<(int)(DSQ / 4 / cb), cb, 0, stream>>>(Wo, Woh, (int)(DSQ / 4));

  const int gblocks = (M_ / 32) * (D_ / 64) / 8;  // 512 blocks x 8 waves
  gemm_bt_f16<<<gblocks, 256, 0, stream>>>(Xq, Wqh, bq, nullptr, Qh,  M_, D_, D_, 1);
  gemm_bt_f16<<<gblocks, 256, 0, stream>>>(Xk, Wkh, bk, nullptr, Khb, M_, D_, D_, 1);
  gemm_bt_f16<<<gblocks, 256, 0, stream>>>(Xv, Wvh, bv, nullptr, Vtb, M_, D_, D_, 2);

  flash_attn_f16<<<dim3(T_ / 64, B_ * H_), 128, 0, stream>>>(Qh, Khb, Vtb, mask, Oh);

  gemm_bt_f16<<<gblocks, 256, 0, stream>>>(Oh, Woh, bo, out, nullptr, M_, D_, D_, 0);
}
